// SCA_41077067219260
// MI455X (gfx1250) — matrix operand order repaired and checked by reading, not compiled, not run on hardware
//
#include <hip/hip_runtime.h>

#define C_IN 512
#define C3   1536
#define HEADS 64
#define CH    8
#define H_IMG 128
#define W_IMG 128
#define HW    (H_IMG * W_IMG)
#define IMGS  4   // img = modality*2 + batch   (mod 0 = hsi, 1 = lidar)

typedef __attribute__((ext_vector_type(16))) __bf16        bf16x16;
typedef __attribute__((ext_vector_type(8)))  float         f32x8;
typedef __attribute__((ext_vector_type(4)))  unsigned int  u32x4;
typedef __attribute__((ext_vector_type(4)))  int           i32x4;

union Frag { bf16x16 b; u32x4 u[2]; };

#if __has_builtin(__builtin_amdgcn_global_load_async_to_lds_b128)
#define HAVE_ASYNC_LDS 1
#else
#define HAVE_ASYNC_LDS 0
#endif

__device__ __forceinline__ unsigned short f2bf(float f) {
    unsigned int u = __float_as_uint(f);
    u += 0x7FFFu + ((u >> 16) & 1u);          // round-to-nearest-even
    return (unsigned short)(u >> 16);
}
__device__ __forceinline__ float bf2f(unsigned short h) {
    return __uint_as_float(((unsigned int)h) << 16);
}

// ---------------------------------------------------------------------------
// Kernel 0: f32 -> bf16 conversion (vectorized), used for X panels + weights
// ---------------------------------------------------------------------------
__global__ __launch_bounds__(256)
void cvt_bf16(const float* __restrict__ src, unsigned short* __restrict__ dst, int n4)
{
    int i = blockIdx.x * 256 + threadIdx.x;
    if (i >= n4) return;
    float4 v = reinterpret_cast<const float4*>(src)[i];
    unsigned long long p =  (unsigned long long)f2bf(v.x)
                         | ((unsigned long long)f2bf(v.y) << 16)
                         | ((unsigned long long)f2bf(v.z) << 32)
                         | ((unsigned long long)f2bf(v.w) << 48);
    reinterpret_cast<unsigned long long*>(dst)[i] = p;
}

// ---------------------------------------------------------------------------
// Kernel 1: 1x1 conv as GEMM  Y[1536,HW] = W[1536,512] * X[512,HW] + bias
// bf16 WMMA. Block tile 64(M) x 256(N), K step 32.
// 8 waves: wave tile 32(M) x 64(N) = 2 A-frags x 4 B-frags = 8 WMMAs/step.
// W tile staged via GLOBAL_LOAD_ASYNC_TO_LDS_B128 when available.
// ---------------------------------------------------------------------------
__global__ __launch_bounds__(256)
void qkv_gemm_wmma(const unsigned short* __restrict__ Xbf,
                   const unsigned short* __restrict__ Wbf,
                   const float* __restrict__ hsi_b, const float* __restrict__ lidar_b,
                   unsigned short* __restrict__ y)
{
    __shared__ __align__(16) unsigned short lW[64 * 40];    // [m][k], pitch 40 halves (80B)
    __shared__ __align__(16) unsigned short lX[256 * 40];   // [n][k] transposed, pitch 40

    const int tid = threadIdx.x;
    const int img = blockIdx.z;
    const int mod = img >> 1;
    const int M0 = blockIdx.y * 64;
    const int N0 = blockIdx.x * 256;

    const unsigned short* Wg = Wbf + (size_t)mod * C3 * C_IN;
    const unsigned short* Xg = Xbf + (size_t)img * C_IN * HW;
    const float* Bg = mod ? lidar_b : hsi_b;
    unsigned short* Yg = y + (size_t)img * C3 * HW;

    const int lane = tid & 31;
    const int wave = tid >> 5;
    const int wm = wave & 1;          // M half: 0/32
    const int wn = wave >> 1;         // N quarter: 0..3 (x64)
    const int lane15 = lane & 15;
    const int kA = (lane >> 4) * 8;   // A-frag K phase (ISA 16-bit A layout)
    const int kB = (lane >> 4) * 8;   // B-frag K phase (same as A)

    const f32x8 vzero = {0.f, 0.f, 0.f, 0.f, 0.f, 0.f, 0.f, 0.f};
    f32x8 acc[2][4];
    #pragma unroll
    for (int i = 0; i < 2; ++i)
        #pragma unroll
        for (int s = 0; s < 4; ++s) acc[i][s] = vzero;

    // per-thread staging coordinates
    const int wr  = tid >> 2;              // W row 0..63
    const int wkk = (tid & 3) * 8;         // W k-chunk 0/8/16/24
    const int xk  = tid >> 5;              // X k row base 0..7
    const int xn8 = (tid & 31) * 8;        // X n chunk (8 halves)

    for (int k0 = 0; k0 < C_IN; k0 += 32) {
        __syncthreads();

        // ---- stage W tile (64x32): one 16B chunk per thread ----
        {
            const unsigned short* gsrc = Wg + (size_t)(M0 + wr) * C_IN + (k0 + wkk);
            unsigned short* ldst = &lW[wr * 40 + wkk];
#if HAVE_ASYNC_LDS
            __builtin_amdgcn_global_load_async_to_lds_b128(
                (__attribute__((address_space(1))) i32x4*)
                    (__attribute__((address_space(1))) void*)(void*)const_cast<unsigned short*>(gsrc),
                (__attribute__((address_space(3))) i32x4*)
                    (__attribute__((address_space(3))) void*)(void*)ldst,
                0, 0);
#else
            *reinterpret_cast<u32x4*>(ldst) = *reinterpret_cast<const u32x4*>(gsrc);
#endif
        }

        // ---- stage X tile transposed [n][k] (32x256): 4x b128 per thread ----
        #pragma unroll
        for (int c = 0; c < 4; ++c) {
            int k = xk + c * 8;
            u32x4 d = *reinterpret_cast<const u32x4*>(Xg + (size_t)(k0 + k) * HW + N0 + xn8);
            #pragma unroll
            for (int j = 0; j < 4; ++j) {
                unsigned int w = d[j];
                lX[(xn8 + 2 * j)     * 40 + k] = (unsigned short)(w & 0xFFFFu);
                lX[(xn8 + 2 * j + 1) * 40 + k] = (unsigned short)(w >> 16);
            }
        }

        if (k0 + 32 < C_IN)   // hint next X panel toward L2 (global_prefetch_b8)
            __builtin_prefetch(Xg + (size_t)(k0 + 32 + xk) * HW + N0 + xn8, 0, 1);

#if HAVE_ASYNC_LDS
 #if __has_builtin(__builtin_amdgcn_s_wait_asynccnt)
        __builtin_amdgcn_s_wait_asynccnt(0);
 #else
        asm volatile("s_wait_asynccnt 0x0" ::: "memory");
 #endif
#endif
        __syncthreads();

        // ---- preload all fragments, then issue 8 WMMAs back-to-back ----
        Frag fa[2], fb[4];
        #pragma unroll
        for (int i = 0; i < 2; ++i) {
            const unsigned short* ap = &lW[(wm * 32 + i * 16 + lane15) * 40];
            fa[i].u[0] = *reinterpret_cast<const u32x4*>(ap + kA);       // K = kA..kA+7
            fa[i].u[1] = *reinterpret_cast<const u32x4*>(ap + 16 + kA);  // K = 16+kA..
        }
        #pragma unroll
        for (int s = 0; s < 4; ++s) {
            const unsigned short* bp = &lX[(wn * 64 + s * 16 + lane15) * 40 + kB];
            fb[s].u[0] = *reinterpret_cast<const u32x4*>(bp);            // K = kB..kB+7
            fb[s].u[1] = *reinterpret_cast<const u32x4*>(bp + 16);       // K = 16+kB..
        }
        #pragma unroll
        for (int i = 0; i < 2; ++i) {
            #pragma unroll
            for (int s = 0; s < 4; ++s) {
                acc[i][s] = __builtin_amdgcn_wmma_f32_16x16x32_bf16(
                    false, fa[i].b, false, fb[s].b, (short)0, acc[i][s], false, false);
            }
        }
    }

    // epilogue: C/D layout — VGPR r: lanes 0-15 -> M=r, lanes 16-31 -> M=8+r
    const int mHi = (lane >> 4) * 8;
    #pragma unroll
    for (int i = 0; i < 2; ++i) {
        #pragma unroll
        for (int s = 0; s < 4; ++s) {
            int col = N0 + wn * 64 + s * 16 + lane15;
            #pragma unroll
            for (int r = 0; r < 8; ++r) {
                int ch = M0 + wm * 32 + i * 16 + mHi + r;
                float v = acc[i][s][r] + Bg[ch];
                Yg[(size_t)ch * HW + col] = f2bf(v);
            }
        }
    }
}

// ---------------------------------------------------------------------------
// Kernel 2: depthwise 3x3 conv + bias (bf16 in / bf16 out)
// ---------------------------------------------------------------------------
__global__ __launch_bounds__(256)
void dwconv3x3(const unsigned short* __restrict__ y,
               const float* __restrict__ hsi_dw_w, const float* __restrict__ hsi_dw_b,
               const float* __restrict__ lidar_dw_w, const float* __restrict__ lidar_dw_b,
               unsigned short* __restrict__ qkv)
{
    const int img = blockIdx.z;
    const int mod = img >> 1;
    const int ch  = blockIdx.y;
    const int n   = blockIdx.x * 256 + threadIdx.x;
    const int hh = n >> 7, ww = n & 127;

    const float* wt = (mod ? lidar_dw_w : hsi_dw_w) + ch * 9;
    const float* bs = (mod ? lidar_dw_b : hsi_dw_b);
    const unsigned short* src = y + (size_t)img * C3 * HW + (size_t)ch * HW;

    float acc = bs[ch];
    #pragma unroll
    for (int dy = 0; dy < 3; ++dy) {
        int yy = hh + dy - 1;
        if (yy >= 0 && yy < H_IMG) {
            #pragma unroll
            for (int dx = 0; dx < 3; ++dx) {
                int xx = ww + dx - 1;
                if (xx >= 0 && xx < W_IMG)
                    acc += bf2f(src[yy * W_IMG + xx]) * wt[dy * 3 + dx];
            }
        }
    }
    qkv[(size_t)img * C3 * HW + (size_t)ch * HW + n] = f2bf(acc);
}

// ---------------------------------------------------------------------------
// Kernel 3: 8x8 Gram over K=16384 + row norms; attn = g/(|q||k|) * temp
// ---------------------------------------------------------------------------
__global__ __launch_bounds__(256)
void gram_attn(const unsigned short* __restrict__ qkv,
               const float* __restrict__ temp1, const float* __restrict__ temp2,
               float* __restrict__ attn1, float* __restrict__ attn2)
{
    const int head  = blockIdx.x;
    const int b     = blockIdx.y;
    const int which = blockIdx.z;
    const int imgQ = (which ? 2 : 0) + b;
    const int imgK = (which ? 0 : 2) + b;
    const unsigned short* Q = qkv + (size_t)imgQ * C3 * HW + (size_t)(head * CH) * HW;
    const unsigned short* K = qkv + (size_t)imgK * C3 * HW + (size_t)(C_IN + head * CH) * HW;

    float g[64], qq[8], kk[8];
    #pragma unroll
    for (int i = 0; i < 64; ++i) g[i] = 0.f;
    #pragma unroll
    for (int i = 0; i < 8; ++i) { qq[i] = 0.f; kk[i] = 0.f; }

    for (int n = threadIdx.x; n < HW; n += 256) {
        float qv[8], kv[8];
        #pragma unroll
        for (int c = 0; c < 8; ++c) qv[c] = bf2f(Q[(size_t)c * HW + n]);
        #pragma unroll
        for (int d = 0; d < 8; ++d) kv[d] = bf2f(K[(size_t)d * HW + n]);
        #pragma unroll
        for (int c = 0; c < 8; ++c) {
            qq[c] += qv[c] * qv[c];
            kk[c] += kv[c] * kv[c];
            #pragma unroll
            for (int d = 0; d < 8; ++d) g[c * 8 + d] += qv[c] * kv[d];
        }
    }

    __shared__ float tot[80];
    if (threadIdx.x < 80) tot[threadIdx.x] = 0.f;
    __syncthreads();

    auto red = [&](float v, int slot) {
        #pragma unroll
        for (int off = 16; off; off >>= 1) v += __shfl_xor(v, off, 32);
        if ((threadIdx.x & 31) == 0) atomicAdd(&tot[slot], v);
    };
    #pragma unroll
    for (int i = 0; i < 64; ++i) red(g[i], i);
    #pragma unroll
    for (int i = 0; i < 8; ++i) { red(qq[i], 64 + i); red(kk[i], 72 + i); }
    __syncthreads();

    if (threadIdx.x < 64) {
        int c = threadIdx.x >> 3, d = threadIdx.x & 7;
        float nq = fmaxf(sqrtf(tot[64 + c]), 1e-12f);
        float nk = fmaxf(sqrtf(tot[72 + d]), 1e-12f);
        float t  = which ? temp2[head] : temp1[head];
        float val = tot[threadIdx.x] / (nq * nk) * t;
        float* dst = which ? attn2 : attn1;
        dst[(size_t)(b * HEADS + head) * 64 + threadIdx.x] = val;
    }
}

// ---------------------------------------------------------------------------
// Kernel 4: proj conv (128->64 across heads) + BN(eval) + ReLU + softmax(d)
// ---------------------------------------------------------------------------
__global__ __launch_bounds__(64)
void fuse_softmax(const float* __restrict__ attn1, const float* __restrict__ attn2,
                  const float* __restrict__ proj_w, const float* __restrict__ proj_b,
                  const float* __restrict__ gamma, const float* __restrict__ beta,
                  const float* __restrict__ mean,  const float* __restrict__ var,
                  float* __restrict__ a1, float* __restrict__ a2)
{
    const int o = blockIdx.x;
    const int b = blockIdx.y;
    const int t = threadIdx.x;

    float s = proj_b[o];
    const float* w = proj_w + o * 128;
    for (int i = 0; i < 64; ++i) s += w[i]      * attn1[(size_t)(b * 64 + i) * 64 + t];
    for (int i = 0; i < 64; ++i) s += w[64 + i] * attn2[(size_t)(b * 64 + i) * 64 + t];
    s = (s - mean[o]) * rsqrtf(var[o] + 1e-5f) * gamma[o] + beta[o];
    s = fmaxf(s, 0.f);

    float z1 = s + attn1[(size_t)(b * 64 + o) * 64 + t];
    float z2 = s + attn2[(size_t)(b * 64 + o) * 64 + t];

    float m1 = z1, m2 = z2;
    #pragma unroll
    for (int off = 4; off; off >>= 1) {
        m1 = fmaxf(m1, __shfl_xor(m1, off, 8));
        m2 = fmaxf(m2, __shfl_xor(m2, off, 8));
    }
    float e1 = expf(z1 - m1), e2 = expf(z2 - m2);
    float s1 = e1, s2 = e2;
    #pragma unroll
    for (int off = 4; off; off >>= 1) {
        s1 += __shfl_xor(s1, off, 8);
        s2 += __shfl_xor(s2, off, 8);
    }
    a1[(size_t)(b * 64 + o) * 64 + t] = e1 / s1;
    a2[(size_t)(b * 64 + o) * 64 + t] = e2 / s2;
}

// ---------------------------------------------------------------------------
// Kernel 5: out[c,n] = sum_d a[c,d]*v[d,n] + v[c,n] + x[c,n]
// ---------------------------------------------------------------------------
__global__ __launch_bounds__(256)
void attn_out(const unsigned short* __restrict__ qkv,
              const float* __restrict__ a1, const float* __restrict__ a2,
              const float* __restrict__ hsi, const float* __restrict__ lidar,
              float* __restrict__ out)
{
    __shared__ float aS[64];
    const int img = blockIdx.z;
    const int mod = img >> 1, b = img & 1;
    const int h = blockIdx.y;
    const int n = blockIdx.x * 256 + threadIdx.x;

    const float* A = (mod ? a2 : a1) + (size_t)(b * 64 + h) * 64;
    if (threadIdx.x < 64) aS[threadIdx.x] = A[threadIdx.x];
    __syncthreads();

    const unsigned short* V = qkv + (size_t)img * C3 * HW + (size_t)(2 * C_IN + h * CH) * HW;
    const float* X = (mod ? lidar : hsi) + (size_t)b * C_IN * HW + (size_t)(h * CH) * HW;
    float* O = out + (size_t)mod * 2 * C_IN * HW + (size_t)b * C_IN * HW + (size_t)(h * CH) * HW;

    float vv[8];
    #pragma unroll
    for (int d = 0; d < 8; ++d) vv[d] = bf2f(V[(size_t)d * HW + n]);
    #pragma unroll
    for (int c = 0; c < 8; ++c) {
        float s = vv[c] + X[(size_t)c * HW + n];
        #pragma unroll
        for (int d = 0; d < 8; ++d) s += aS[c * 8 + d] * vv[d];
        O[(size_t)c * HW + n] = s;
    }
}

// ---------------------------------------------------------------------------
extern "C" void kernel_launch(void* const* d_in, const int* in_sizes, int n_in,
                              void* d_out, int out_size, void* d_ws, size_t ws_size,
                              hipStream_t stream)
{
    const float* hsi         = (const float*)d_in[0];
    const float* lidar       = (const float*)d_in[1];
    const float* hsi_qkv_w   = (const float*)d_in[2];
    const float* hsi_qkv_b   = (const float*)d_in[3];
    const float* lidar_qkv_w = (const float*)d_in[4];
    const float* lidar_qkv_b = (const float*)d_in[5];
    const float* hsi_dw_w    = (const float*)d_in[6];
    const float* hsi_dw_b    = (const float*)d_in[7];
    const float* lidar_dw_w  = (const float*)d_in[8];
    const float* lidar_dw_b  = (const float*)d_in[9];
    const float* temp1       = (const float*)d_in[10];
    const float* temp2       = (const float*)d_in[11];
    const float* proj_w      = (const float*)d_in[12];
    const float* proj_b      = (const float*)d_in[13];
    const float* bn_gamma    = (const float*)d_in[14];
    const float* bn_beta     = (const float*)d_in[15];
    const float* bn_mean     = (const float*)d_in[16];
    const float* bn_var      = (const float*)d_in[17];
    float* out = (float*)d_out;

    // workspace layout (~473 MB): bf16 intermediates halve HBM traffic;
    // bf16 X panels (16 MB/batch) are L2-resident across the 24 M-blocks.
    char* ws = (char*)d_ws;
    const size_t ybytes = (size_t)IMGS * C3 * HW * sizeof(unsigned short); // 201 MB
    unsigned short* y   = (unsigned short*)ws;
    unsigned short* qkv = (unsigned short*)(ws + ybytes);
    float* attn1 = (float*)(ws + 2 * ybytes);
    float* attn2 = attn1 + 2 * HEADS * 64;
    float* a1    = attn2 + 2 * HEADS * 64;
    float* a2    = a1    + 2 * HEADS * 64;
    unsigned short* Xbf = (unsigned short*)(a2 + 2 * HEADS * 64);
    unsigned short* Wbf = Xbf + (size_t)IMGS * C_IN * HW;

    // convert activations + 1x1 weights to bf16
    {
        const int nxa = 2 * C_IN * HW;       // one modality (both batches)
        const int nw  = C3 * C_IN;           // one modality's weights
        cvt_bf16<<<dim3((nxa / 4 + 255) / 256), 256, 0, stream>>>(hsi,   Xbf,                nxa / 4);
        cvt_bf16<<<dim3((nxa / 4 + 255) / 256), 256, 0, stream>>>(lidar, Xbf + (size_t)nxa,  nxa / 4);
        cvt_bf16<<<dim3((nw  / 4 + 255) / 256), 256, 0, stream>>>(hsi_qkv_w,   Wbf,               nw / 4);
        cvt_bf16<<<dim3((nw  / 4 + 255) / 256), 256, 0, stream>>>(lidar_qkv_w, Wbf + (size_t)nw,  nw / 4);
    }

    qkv_gemm_wmma<<<dim3(HW / 256, C3 / 64, IMGS), 256, 0, stream>>>(
        Xbf, Wbf, hsi_qkv_b, lidar_qkv_b, y);

    dwconv3x3<<<dim3(HW / 256, C3, IMGS), 256, 0, stream>>>(
        y, hsi_dw_w, hsi_dw_b, lidar_dw_w, lidar_dw_b, qkv);

    gram_attn<<<dim3(HEADS, 2, 2), 256, 0, stream>>>(qkv, temp1, temp2, attn1, attn2);

    fuse_softmax<<<dim3(HEADS, 2), 64, 0, stream>>>(
        attn1, attn2, proj_w, proj_b, bn_gamma, bn_beta, bn_mean, bn_var, a1, a2);

    attn_out<<<dim3(HW / 256, HEADS, IMGS), 256, 0, stream>>>(qkv, a1, a2, hsi, lidar, out);
}
